// H3Expand_56195352101230
// MI455X (gfx1250) — matrix operand order repaired and checked by reading, not compiled, not run on hardware
//
#include <hip/hip_runtime.h>
#include <stdint.h>

typedef __attribute__((ext_vector_type(16))) _Float16 v16h;
typedef __attribute__((ext_vector_type(8)))  float    v8f;

#define B_SZ    8
#define L_SEQ   2048
#define D_MODEL 1024
#define N_STATE 64
#define N_REC   8
#define LC      256

// ---------------------------------------------------------------------------
// Prep: split f32 -> (hi, lo) f16 planes.  x = hi + lo recovers ~22-bit
// mantissa, so 3 WMMAs (hi*hi + lo*hi + hi*lo) give fp32-class GEMM accuracy.
// ---------------------------------------------------------------------------
__global__ __launch_bounds__(256) void split_hl(const float* __restrict__ in,
                                                _Float16* __restrict__ hi,
                                                _Float16* __restrict__ lo, int n) {
  int i = blockIdx.x * 256 + threadIdx.x;
  const int stride = gridDim.x * 256;
  for (; i < n; i += stride) {
    const float x = in[i];
    const _Float16 h = (_Float16)x;
    hi[i] = h;
    lo[i] = (_Float16)(x - (float)h);
  }
}

// ---------------------------------------------------------------------------
// WMMA 16x16x32 f16 fragment layouts (ISA 7.12.2, wave32):
// A (MxK=16x32): lane {half=lane>>4, m=lane&15}; elem j<8 -> K=half*8+j,
//                elem j>=8 -> K=16+half*8+(j-8).
// B (KxN=32x16): lane {half, n=lane&15}; same K mapping as A.
// ---------------------------------------------------------------------------
__device__ inline v16h frag_A(const _Float16* __restrict__ p, int lane, int ld) {
  v16h f;
  const int m = lane & 15, ks = (lane >> 4) * 8;
  const _Float16* q = p + (size_t)m * ld + ks;
#pragma unroll
  for (int j = 0; j < 8; ++j) f[j] = q[j];
#pragma unroll
  for (int j = 0; j < 8; ++j) f[8 + j] = q[16 + j];
  return f;
}
__device__ inline v16h frag_B(const _Float16* __restrict__ p, int lane, int ld) {
  v16h f;
  const int nn = lane & 15, ks = (lane >> 4) * 8;
  const _Float16* q = p + (size_t)nn * ld + ks;
#pragma unroll
  for (int j = 0; j < 16; ++j) f[j] = q[j < 8 ? j : j + 8];
  return f;
}

// ---------------------------------------------------------------------------
// K1: k0[b,h,l] = sum_d W[h,d]*u[b,l,d] + bias[h]
// Wave tile 32(M)x64(N); 8 waves as 4(M)x2(N) -> 128x128 per WG.
// ---------------------------------------------------------------------------
__global__ __launch_bounds__(256) void h3_gemm_wmma(
    const _Float16* __restrict__ uhi, const _Float16* __restrict__ ulo,
    const _Float16* __restrict__ Whi, const _Float16* __restrict__ Wlo,
    const float* __restrict__ bias, float* __restrict__ k0) {
  const int lane = threadIdx.x & 31;
  const int wave = threadIdx.x >> 5;
  const int wm   = wave >> 1;                 // 0..3
  const int wn   = wave & 1;                  // 0..1
  const int b    = blockIdx.z;
  const int m0   = blockIdx.y * 128 + wm * 32;
  const int l0   = blockIdx.x * 128 + wn * 64;

  const _Float16* __restrict__ uh = uhi + (size_t)b * L_SEQ * D_MODEL;
  const _Float16* __restrict__ ul = ulo + (size_t)b * L_SEQ * D_MODEL;

  v8f acc[2][4];
#pragma unroll
  for (int i = 0; i < 2; ++i)
#pragma unroll
    for (int j = 0; j < 4; ++j)
#pragma unroll
      for (int v = 0; v < 8; ++v) acc[i][j][v] = 0.0f;

  for (int k = 0; k < D_MODEL; k += 32) {
    if (k + 32 < D_MODEL) {                       // gfx1250 global_prefetch_b8
      __builtin_prefetch(Whi + (size_t)m0 * D_MODEL + k + 32, 0, 1);
      __builtin_prefetch(uh + (size_t)l0 * D_MODEL + k + 32, 0, 1);
    }
    v16h aH[2], aL[2], bH[4], bL[4];
#pragma unroll
    for (int i = 0; i < 2; ++i) {
      aH[i] = frag_A(Whi + (size_t)(m0 + 16 * i) * D_MODEL + k, lane, D_MODEL);
      aL[i] = frag_A(Wlo + (size_t)(m0 + 16 * i) * D_MODEL + k, lane, D_MODEL);
    }
#pragma unroll
    for (int j = 0; j < 4; ++j) {
      bH[j] = frag_B(uh + (size_t)(l0 + 16 * j) * D_MODEL + k, lane, D_MODEL);
      bL[j] = frag_B(ul + (size_t)(l0 + 16 * j) * D_MODEL + k, lane, D_MODEL);
    }
    // three passes over all 8 accumulators -> long separation on D->C chains
#pragma unroll
    for (int i = 0; i < 2; ++i)
#pragma unroll
      for (int j = 0; j < 4; ++j)
        acc[i][j] = __builtin_amdgcn_wmma_f32_16x16x32_f16(
            false, aH[i], false, bH[j], (short)0, acc[i][j], false, false);
#pragma unroll
    for (int i = 0; i < 2; ++i)
#pragma unroll
      for (int j = 0; j < 4; ++j)
        acc[i][j] = __builtin_amdgcn_wmma_f32_16x16x32_f16(
            false, aL[i], false, bH[j], (short)0, acc[i][j], false, false);
#pragma unroll
    for (int i = 0; i < 2; ++i)
#pragma unroll
      for (int j = 0; j < 4; ++j)
        acc[i][j] = __builtin_amdgcn_wmma_f32_16x16x32_f16(
            false, aH[i], false, bL[j], (short)0, acc[i][j], false, false);
  }

  const int half = lane >> 4, col = lane & 15;
#pragma unroll
  for (int i = 0; i < 2; ++i)
#pragma unroll
    for (int j = 0; j < 4; ++j) {
      const int l = l0 + j * 16 + col;
#pragma unroll
      for (int v = 0; v < 8; ++v) {
        const int hrow = m0 + i * 16 + v + 8 * half;
        k0[((size_t)b * D_MODEL + hrow) * L_SEQ + l] = acc[i][j][v] + bias[hrow];
      }
    }
}

__device__ inline void cxmul(float& xr, float& xi, float yr, float yi) {
  const float t = xr * yr - xi * yi;
  xi = xr * yi + xi * yr;
  xr = t;
}

// Async global->LDS staging of one 2048-float row at dst+64 (two 4KB passes).
// The 24-bit IOFFSET is added to BOTH the LDS and global addresses (ISA 4.4),
// so one base pair + offset:4096 covers the second pass.  Tracked by ASYNCcnt.
__device__ inline void stage_row_async(const float* __restrict__ gsrc,
                                       float* __restrict__ dst, int tid) {
  const uint32_t lds0 = (uint32_t)(uintptr_t)(dst + 64) + (uint32_t)tid * 16u;
  const uint64_t g0   = (uint64_t)(uintptr_t)(gsrc + tid * 4);
  asm volatile("global_load_async_to_lds_b128 %0, %1, off"
               :: "v"(lds0), "v"(g0) : "memory");
  asm volatile("global_load_async_to_lds_b128 %0, %1, off offset:4096"
               :: "v"(lds0), "v"(g0) : "memory");
}

// ---------------------------------------------------------------------------
// K2: per-h fused FIR + WMMA state expansion.  256 threads per WG, one WG per h.
//  contrib[n,(b,r)] = sum_{l<256} P[n,l] * kc[b, r*256+l],  P = dA^(255-l)*dB
//  A-fragment = P kept in registers (f32), stepped by *dA^32 per K-step,
//  emitted as hi/lo f16.  Row staging is double-buffered async DMA (ASYNCcnt),
//  overlapping the b+1 row copy with the FIR of row b.
// ---------------------------------------------------------------------------
__global__ __launch_bounds__(256) void h3_expand_wmma(
    const float* __restrict__ k0,
    const float* __restrict__ shiftB, const float* __restrict__ shiftC,
    const float* __restrict__ ssmD, const float* __restrict__ log_dt,
    const float* __restrict__ log_A_real, const float* __restrict__ A_imag,
    const float* __restrict__ B_re, const float* __restrict__ B_im,
    float* __restrict__ out) {
  // 32KB pool: f16 kc[b][l] during FIR+matmul, then f32 contrib[n][col][2]
  __shared__ __align__(16) unsigned char pool[B_SZ * L_SEQ * 2];
  _Float16* kc16 = (_Float16*)pool;
  float*    ctb  = (float*)pool;
  __shared__ __align__(16) float rawbuf[2][L_SEQ + 64];
  __shared__ float taps[N_STATE];
  __shared__ float sBB[N_STATE], sCC[N_STATE];

  const int tid = threadIdx.x;
  const int h   = blockIdx.x;

  // causal zero-pad history (DMA never writes slots 0..63)
  if (tid < 128) rawbuf[tid >> 6][tid & 63] = 0.0f;
  if (tid < N_STATE) {
    sBB[tid] = shiftB[h * N_STATE + tid];
    sCC[tid] = shiftC[h * N_STATE + tid];
  }
  // kick off DMA of row b=0 while taps are computed
  stage_row_async(k0 + ((size_t)0 * D_MODEL + h) * L_SEQ, rawbuf[0], tid);
  __syncthreads();
  if (tid < N_STATE) {   // Ks[h,j] = sum_{m=0}^{63-j} C[h,m+j]*B[h,m]
    float t = 0.0f;
    for (int m = 0; m + tid < N_STATE; ++m) t = fmaf(sCC[m + tid], sBB[m], t);
    taps[tid] = t;
  }
  const float dt = expf(log_dt[h]);
  const float Dh = ssmD[h];

  // ---- FIR (64-tap causal) + D skip, f16 result into LDS ----
  for (int b = 0; b < B_SZ; ++b) {
    if (b + 1 < B_SZ)      // overlap next row's DMA with this row's FIR
      stage_row_async(k0 + ((size_t)(b + 1) * D_MODEL + h) * L_SEQ,
                      rawbuf[(b + 1) & 1], tid);
    // async loads complete in order: <=2 outstanding leaves only next row's ops
    if (b + 1 < B_SZ) { asm volatile("s_wait_asynccnt 2" ::: "memory"); }
    else              { asm volatile("s_wait_asynccnt 0" ::: "memory"); }
    __syncthreads();
    const float* __restrict__ rw = rawbuf[b & 1];
#pragma unroll
    for (int q = 0; q < 8; ++q) {
      const int l = tid + 256 * q;
      float a = Dh * rw[64 + l];
#pragma unroll 16
      for (int j = 0; j < N_STATE; ++j) a = fmaf(taps[j], rw[64 + l - j], a);
      kc16[b * L_SEQ + l] = (_Float16)a;
    }
    __syncthreads();     // FIR readers done before this buffer is re-DMA'd
  }

  // ---- per-wave setup: wave = mt(0..3) x nh(0..1); lane row n = mt*16+(lane&15)
  const int lane = tid & 31;
  const int wave = tid >> 5;
  const int mt = wave & 3;
  const int nh = wave >> 2;
  const int n  = mt * 16 + (lane & 15);

  const float Ar = -expf(log_A_real[h * N_STATE + n]);
  const float Ai = A_imag[h * N_STATE + n];
  const float eA  = expf(dt * Ar);
  const float dAr = eA * cosf(dt * Ai);           // |dt*Ai| <= ~20 rad: safe
  const float dAi = eA * sinf(dt * Ai);
  const float den = Ar * Ar + Ai * Ai;            // >= 0.25
  const float nr = dAr - 1.0f, ni = dAi;
  const float wqr = (nr * Ar + ni * Ai) / den;
  const float wqi = (ni * Ar - nr * Ai) / den;
  const float dBr = B_re[h * N_STATE + n] * wqr - B_im[h * N_STATE + n] * wqi;
  const float dBi = B_re[h * N_STATE + n] * wqi + B_im[h * N_STATE + n] * wqr;

  // dA^(2^t) by complex squaring (no large-angle trig)
  float p2r[6], p2i[6];
  p2r[0] = dAr; p2i[0] = dAi;
#pragma unroll
  for (int t = 1; t < 6; ++t) {
    p2r[t] = p2r[t - 1]; p2i[t] = p2i[t - 1];
    cxmul(p2r[t], p2i[t], p2r[t - 1], p2i[t - 1]);
  }
  const float A32r = p2r[5], A32i = p2i[5];

  // init P for K-step s=7 (l=224..255): P[elem] = dA^(31-ll)*dB, binary powers
  const int h8 = (lane >> 4) * 8;
  float Pre[16], Pim[16];
#pragma unroll
  for (int j = 0; j < 16; ++j) {
    const int ll = (j < 8) ? (h8 + j) : (16 + h8 + (j - 8));
    const int ex = 31 - ll;
    float xr = 1.0f, xi = 0.0f;
#pragma unroll
    for (int t = 0; t < 5; ++t)
      if ((ex >> t) & 1) cxmul(xr, xi, p2r[t], p2i[t]);
    Pre[j] = xr * dBr - xi * dBi;
    Pim[j] = xr * dBi + xi * dBr;
  }

  v8f accR[2], accI[2];
#pragma unroll
  for (int j = 0; j < 2; ++j)
#pragma unroll
    for (int v = 0; v < 8; ++v) { accR[j][v] = 0.0f; accI[j][v] = 0.0f; }

  for (int s = 7; s >= 0; --s) {
    v16h pRH, pRL, pIH, pIL;
#pragma unroll
    for (int j = 0; j < 16; ++j) {
      const _Float16 h1 = (_Float16)Pre[j];
      pRH[j] = h1; pRL[j] = (_Float16)(Pre[j] - (float)h1);
      const _Float16 h2 = (_Float16)Pim[j];
      pIH[j] = h2; pIL[j] = (_Float16)(Pim[j] - (float)h2);
    }
    v16h bf[2];
#pragma unroll
    for (int jt = 0; jt < 2; ++jt) {
      const int col  = (nh * 2 + jt) * 16 + (lane & 15);
      const int bcol = col >> 3, rcol = col & 7;
      const _Float16* q = kc16 + bcol * L_SEQ + rcol * LC + s * 32 + (lane >> 4) * 8;
#pragma unroll
      for (int j = 0; j < 16; ++j) bf[jt][j] = q[j < 8 ? j : j + 8];
    }
    accR[0] = __builtin_amdgcn_wmma_f32_16x16x32_f16(false, pRH, false, bf[0], (short)0, accR[0], false, false);
    accR[1] = __builtin_amdgcn_wmma_f32_16x16x32_f16(false, pRH, false, bf[1], (short)0, accR[1], false, false);
    accI[0] = __builtin_amdgcn_wmma_f32_16x16x32_f16(false, pIH, false, bf[0], (short)0, accI[0], false, false);
    accI[1] = __builtin_amdgcn_wmma_f32_16x16x32_f16(false, pIH, false, bf[1], (short)0, accI[1], false, false);
    accR[0] = __builtin_amdgcn_wmma_f32_16x16x32_f16(false, pRL, false, bf[0], (short)0, accR[0], false, false);
    accR[1] = __builtin_amdgcn_wmma_f32_16x16x32_f16(false, pRL, false, bf[1], (short)0, accR[1], false, false);
    accI[0] = __builtin_amdgcn_wmma_f32_16x16x32_f16(false, pIL, false, bf[0], (short)0, accI[0], false, false);
    accI[1] = __builtin_amdgcn_wmma_f32_16x16x32_f16(false, pIL, false, bf[1], (short)0, accI[1], false, false);
    if (s > 0) {
#pragma unroll
      for (int j = 0; j < 16; ++j) cxmul(Pre[j], Pim[j], A32r, A32i);   // *= dA^32
    }
  }
  __syncthreads();   // all kc16 reads done; pool becomes contrib

  const int half = lane >> 4, colb = lane & 15;
#pragma unroll
  for (int jt = 0; jt < 2; ++jt) {
    const int col = (nh * 2 + jt) * 16 + colb;
#pragma unroll
    for (int v = 0; v < 8; ++v) {
      const int nn = mt * 16 + v + 8 * half;
      ctb[(nn * 64 + col) * 2 + 0] = accR[jt][v];
      ctb[(nn * 64 + col) * 2 + 1] = accI[jt][v];
    }
  }
  __syncthreads();

  // ---- cross-chunk scan: x_r = dA^256 * x_{r-1} + contrib[r], emit Re(x) ----
  for (int p = tid; p < 512; p += 256) {
    const int n2 = p >> 3;
    const int b2 = p & 7;
    const float Ar2 = -expf(log_A_real[h * N_STATE + n2]);
    const float Ai2 = A_imag[h * N_STATE + n2];
    const float e2  = expf(dt * Ar2);
    float Sr = e2 * cosf(dt * Ai2);
    float Si = e2 * sinf(dt * Ai2);
#pragma unroll
    for (int t = 0; t < 8; ++t) cxmul(Sr, Si, Sr, Si);   // dA -> dA^256
    float xr = 0.0f, xi = 0.0f;
#pragma unroll
    for (int r = 0; r < N_REC; ++r) {
      const float cr = ctb[(n2 * 64 + b2 * 8 + r) * 2 + 0];
      const float ci = ctb[(n2 * 64 + b2 * 8 + r) * 2 + 1];
      const float nxr = Sr * xr - Si * xi + cr;
      const float nxi = Sr * xi + Si * xr + ci;
      xr = nxr; xi = nxi;
      out[((size_t)b2 * N_REC + r) * (D_MODEL * N_STATE) + (size_t)h * N_STATE + n2] = xr;
    }
  }
}

extern "C" void kernel_launch(void* const* d_in, const int* in_sizes, int n_in,
                              void* d_out, int out_size, void* d_ws, size_t ws_size,
                              hipStream_t stream) {
  const float* u    = (const float*)d_in[0];
  const float* W    = (const float*)d_in[1];
  const float* bias = (const float*)d_in[2];
  const float* shB  = (const float*)d_in[3];
  const float* shC  = (const float*)d_in[4];
  const float* sD   = (const float*)d_in[5];
  const float* ldt  = (const float*)d_in[6];
  const float* lar  = (const float*)d_in[7];
  const float* aim  = (const float*)d_in[8];
  const float* bre  = (const float*)d_in[9];
  const float* bim  = (const float*)d_in[10];
  float* out = (float*)d_out;

  // workspace layout: [k0 f32 64MB][Whi 2MB][Wlo 2MB][uhi 32MB][ulo 32MB]
  float*    k0  = (float*)d_ws;
  _Float16* Whi = (_Float16*)((char*)d_ws + (size_t)64 * 1024 * 1024);
  _Float16* Wlo = Whi + (size_t)D_MODEL * D_MODEL;
  _Float16* uhi = Wlo + (size_t)D_MODEL * D_MODEL;
  _Float16* ulo = uhi + (size_t)B_SZ * L_SEQ * D_MODEL;

  const int nW = D_MODEL * D_MODEL;
  const int nU = B_SZ * L_SEQ * D_MODEL;
  split_hl<<<512, 256, 0, stream>>>(W, Whi, Wlo, nW);
  split_hl<<<2048, 256, 0, stream>>>(u, uhi, ulo, nU);

  dim3 g1(L_SEQ / 128, D_MODEL / 128, B_SZ);   // (16, 8, 8)
  h3_gemm_wmma<<<g1, 256, 0, stream>>>(uhi, ulo, Whi, Wlo, bias, k0);

  h3_expand_wmma<<<dim3(D_MODEL), 256, 0, stream>>>(
      k0, shB, shC, sD, ldt, lar, aim, bre, bim, out);

  (void)in_sizes; (void)n_in; (void)out_size; (void)ws_size;
}
